// SeNER_79087527788640
// MI455X (gfx1250) — matrix operand order repaired and checked by reading, not compiled, not run on hardware
//
#include <hip/hip_runtime.h>
#include <hip/hip_bf16.h>

typedef __attribute__((ext_vector_type(16))) _Float16 v16h;
typedef __attribute__((ext_vector_type(8)))  _Float16 v8h;
typedef __attribute__((ext_vector_type(4)))  _Float16 v4h;
typedef __attribute__((ext_vector_type(8)))  float    v8f;

#define WMMA_F16(a,b,c) __builtin_amdgcn_wmma_f32_16x16x32_f16(false,(a),false,(b),(short)0,(c),false,false)

union F16x16 { v16h v; v8h h[2]; };

// Load a 16-half fragment as two 16-byte vector loads (both pointers 16B-aligned).
static __device__ __forceinline__ v16h ld_frag2(const _Float16* p0, const _Float16* p1) {
  F16x16 u;
  u.h[0] = *(const v8h*)p0;
  u.h[1] = *(const v8h*)p1;
  return u.v;
}

static constexpr int Bc = 2, Sc = 4096, Hc = 512, NHc = 8, HDc = 64, Wc_ = 128, Lc = 17;

// ---------------------------------------------------------------- cast (4 elems / thread)
__global__ void cast_f32_f16(const float* __restrict__ src, _Float16* __restrict__ dst, int n) {
  int i = (blockIdx.x * 256 + threadIdx.x) * 4;
  if (i + 3 < n) {
    float4 f = *(const float4*)(src + i);
    v4h o = { (_Float16)f.x, (_Float16)f.y, (_Float16)f.z, (_Float16)f.w };
    *(v4h*)(dst + i) = o;
  } else {
    for (int j = i; j < n; ++j) dst[j] = (_Float16)src[j];
  }
}

// ---------------------------------------------------------------- generic WMMA GEMM
// C[M,N] = A[M,K] (f16, lda) @ Bw[K,N] (f16, ldb) + bias ; optional relu.
// B tile staged TRANSPOSED in LDS so B-fragments are contiguous 32-byte runs.
// Requires K % 32 == 0 (true for all call sites: K in {512, 1024}).
#define BM 128
#define BN 64
#define BK 32
#define LDA_S 40   // halves; 80 B row stride (16B aligned)
#define LDBT 40    // halves; transposed B tile row stride

__global__ __launch_bounds__(256)
void gemm_f16(const _Float16* __restrict__ A, int lda,
              const _Float16* __restrict__ Bw, int ldb,
              const float* __restrict__ bias,
              _Float16* __restrict__ outH, float* __restrict__ outF,
              int M, int N, int K, int outLd, int colOff, int relu)
{
  __shared__ _Float16 sA[BM * LDA_S];
  __shared__ _Float16 sBt[BN * LDBT];      // transposed: [n][k]
  const int bm = blockIdx.x * BM;
  const int bn = blockIdx.y * BN;
  const int tid = threadIdx.x;
  const int lane = tid & 31, wid = tid >> 5;
  const int wm = wid & 3, wn = wid >> 2;   // 4 x 2 wave grid
  const int g = lane >> 4, ln = lane & 15;
  const bool bVec = (bn + BN <= N) && ((ldb & 7) == 0);  // vectorized B staging legal?

  v8f c[2][2] = {};

  for (int k0 = 0; k0 < K; k0 += BK) {
    // A tile: 128x32 halves as 512 8-half chunks (16B vector loads/stores)
    for (int id = tid; id < BM * (BK / 8); id += 256) {
      int r = id >> 2, c8 = (id & 3) * 8;
      int gr = bm + r;
      v8h val = {};
      if (gr < M) val = *(const v8h*)(A + (size_t)gr * lda + k0 + c8);
      *(v8h*)(&sA[r * LDA_S + c8]) = val;
      if (gr < M && k0 + BK < K)
        __builtin_prefetch(A + (size_t)gr * lda + k0 + BK + c8, 0, 1);
    }
    // B tile: 32x64 -> transposed LDS. Fast path: one v8h global load per
    // thread (coalesced b128), 8 scalar transposed ds stores.
    if (bVec) {
      int r = tid >> 3, c8 = (tid & 7) * 8;             // 256 chunks exactly
      v8h val = *(const v8h*)(Bw + (size_t)(k0 + r) * ldb + bn + c8);
      #pragma unroll
      for (int j = 0; j < 8; ++j) sBt[(c8 + j) * LDBT + r] = val[j];
    } else {
      for (int i = tid; i < BK * BN; i += 256) {
        int r = i >> 6, cc = i & 63;
        int gc = bn + cc;
        sBt[cc * LDBT + r] = (gc < N) ? Bw[(size_t)(k0 + r) * ldb + gc] : (_Float16)0.f;
      }
    }
    __syncthreads();

    for (int fm = 0; fm < 2; ++fm) {
      const int row = wm * 32 + fm * 16 + ln;
      v16h a = ld_frag2(&sA[row * LDA_S + 16 * g], &sA[row * LDA_S + 16 * g + 8]);
      for (int fn = 0; fn < 2; ++fn) {
        const int col = wn * 32 + fn * 16 + ln;
        v16h bf = ld_frag2(&sBt[col * LDBT + 16 * g], &sBt[col * LDBT + 16 * g + 8]);
        c[fm][fn] = WMMA_F16(a, bf, c[fm][fn]);
      }
    }
    __syncthreads();
  }

  for (int fm = 0; fm < 2; ++fm)
    for (int fn = 0; fn < 2; ++fn) {
      int col = bn + wn * 32 + fn * 16 + ln;
      if (col >= N) continue;
      float bv = bias ? bias[col] : 0.f;
      #pragma unroll
      for (int r = 0; r < 8; ++r) {
        int row = bm + wm * 32 + fm * 16 + r + 8 * g;
        if (row >= M) continue;
        float val = c[fm][fn][r] + bv;
        if (relu) val = val > 0.f ? val : 0.f;
        size_t idx = (size_t)row * outLd + colOff + col;
        if (outF) outF[idx] = val;
        if (outH) outH[idx] = (_Float16)val;
      }
    }
}

// ---------------------------------------------------------------- windowed (banded) attention
// qkv: [B*S, 1536] f16 (q|k|v). One workgroup per (block n, head h, batch b).
// sK[key][d] (contraction-contiguous for QK^T); sVt[d][key] (contraction-contiguous for PV).
#define CW 384
#define KST 72     // halves; 144 B (16B aligned)
#define VST 392    // halves; 784 B (16B aligned)
#define PST 40

__global__ __launch_bounds__(256)
void win_attn(const _Float16* __restrict__ qkv, _Float16* __restrict__ out, int minKey)
{
  __shared__ _Float16 sK[CW * KST];
  __shared__ _Float16 sVt[HDc * VST];
  __shared__ _Float16 sP[8 * 16 * PST];

  const int n = blockIdx.x, h = blockIdx.y, b = blockIdx.z;
  const int tid = threadIdx.x, lane = tid & 31, wid = tid >> 5;
  const int g = lane >> 4, ln = lane & 15;
  const int kbase = n * Wc_ - Wc_;

  // Stage K (row layout, vector stores) and V (transposed, scalar scatter).
  for (int id = tid; id < CW * (HDc / 8); id += 256) {
    int r = id >> 3, d8 = (id & 7) * 8;
    int kp = kbase + r;
    v8h kv = {}, vv = {};
    if (kp >= 0 && kp < Sc) {
      size_t off = ((size_t)(b * Sc + kp)) * (3 * Hc) + h * HDc + d8;
      kv = *(const v8h*)(qkv + off + Hc);
      vv = *(const v8h*)(qkv + off + 2 * Hc);
    }
    *(v8h*)(&sK[r * KST + d8]) = kv;
    #pragma unroll
    for (int j = 0; j < 8; ++j) sVt[(d8 + j) * VST + r] = vv[j];
  }
  __syncthreads();

  const int q0 = wid * 16;
  v16h aq[2];
  {
    const _Float16* qp = qkv + ((size_t)(b * Sc + n * Wc_ + q0 + ln)) * (3 * Hc) + h * HDc;
    #pragma unroll
    for (int ks = 0; ks < 2; ++ks)
      aq[ks] = ld_frag2(qp + ks * 32 + 16 * g, qp + ks * 32 + 16 * g + 8);
  }

  v8f o[4] = {};
  float m[8], l[8];
  #pragma unroll
  for (int r = 0; r < 8; ++r) { m[r] = -3.0e38f; l[r] = 0.f; }
  _Float16* myP = sP + wid * 16 * PST;
  const float scale = 0.125f;   // 1/sqrt(64)

  for (int kt = 0; kt < 12; ++kt) {
    v8f s2[2] = {};
    for (int nt = 0; nt < 2; ++nt) {
      const int col = kt * 32 + nt * 16 + ln;
      #pragma unroll
      for (int ks = 0; ks < 2; ++ks) {
        v16h bf = ld_frag2(&sK[col * KST + ks * 32 + 16 * g],
                           &sK[col * KST + ks * 32 + 16 * g + 8]);
        s2[nt] = WMMA_F16(aq[ks], bf, s2[nt]);
      }
    }
    // scale + band mask
    for (int nt = 0; nt < 2; ++nt) {
      int kp = kbase + kt * 32 + nt * 16 + ln;
      #pragma unroll
      for (int r = 0; r < 8; ++r) {
        int qp = n * Wc_ + q0 + r + 8 * g;
        int rel = kp - qp;
        bool valid = (rel <= Wc_) && (rel >= -Wc_) && (kp >= minKey) && (kp < Sc);
        s2[nt][r] = valid ? s2[nt][r] * scale : -1.0e9f;
      }
    }
    // online softmax (rows live across 16-lane half-groups)
    #pragma unroll
    for (int r = 0; r < 8; ++r) {
      float mx = fmaxf(s2[0][r], s2[1][r]);
      for (int off = 1; off < 16; off <<= 1) mx = fmaxf(mx, __shfl_xor(mx, off, 32));
      float mn = fmaxf(m[r], mx);
      float alpha = __expf(m[r] - mn);
      m[r] = mn;
      float p0 = __expf(s2[0][r] - mn);
      float p1 = __expf(s2[1][r] - mn);
      s2[0][r] = p0; s2[1][r] = p1;
      float ps = p0 + p1;
      for (int off = 1; off < 16; off <<= 1) ps += __shfl_xor(ps, off, 32);
      l[r] = l[r] * alpha + ps;
      #pragma unroll
      for (int f = 0; f < 4; ++f) o[f][r] *= alpha;
    }
    // relayout P: C-fragment layout -> A-fragment layout via per-wave LDS tile
    for (int nt = 0; nt < 2; ++nt)
      #pragma unroll
      for (int r = 0; r < 8; ++r)
        myP[(r + 8 * g) * PST + nt * 16 + ln] = (_Float16)s2[nt][r];
    asm volatile("s_wait_dscnt 0x0" ::: "memory");
    v16h ap = ld_frag2(&myP[ln * PST + 16 * g], &myP[ln * PST + 16 * g + 8]);
    // O += P @ V
    for (int f = 0; f < 4; ++f) {
      v16h bv = ld_frag2(&sVt[(f * 16 + ln) * VST + kt * 32 + 16 * g],
                         &sVt[(f * 16 + ln) * VST + kt * 32 + 16 * g + 8]);
      o[f] = WMMA_F16(ap, bv, o[f]);
    }
  }

  for (int f = 0; f < 4; ++f)
    #pragma unroll
    for (int r = 0; r < 8; ++r) {
      float val = (l[r] > 0.f) ? o[f][r] / l[r] : 0.f;
      size_t row = (size_t)(b * Sc + n * Wc_ + q0 + r + 8 * g);
      out[row * Hc + h * HDc + f * 16 + ln] = (_Float16)val;
    }
}

// ---------------------------------------------------------------- cls-token attention (tiny)
__global__ __launch_bounds__(256)
void cls_attn(const _Float16* __restrict__ qkv, _Float16* __restrict__ out)
{
  __shared__ float sw[Sc];
  __shared__ float red[256];
  __shared__ float sq[HDc];
  const int h = blockIdx.x, b = blockIdx.y;
  const int tid = threadIdx.x;
  if (tid < HDc) sq[tid] = (float)qkv[((size_t)(b * Sc)) * (3 * Hc) + h * HDc + tid];
  __syncthreads();

  float lmax = -3.0e38f;
  for (int s = tid; s < Sc; s += 256) {
    const _Float16* kp = qkv + ((size_t)(b * Sc + s)) * (3 * Hc) + Hc + h * HDc;
    float d = 0.f;
    for (int j = 0; j < HDc; ++j) d += sq[j] * (float)kp[j];
    d *= 0.125f;
    sw[s] = d;
    lmax = fmaxf(lmax, d);
  }
  red[tid] = lmax; __syncthreads();
  for (int st = 128; st > 0; st >>= 1) { if (tid < st) red[tid] = fmaxf(red[tid], red[tid + st]); __syncthreads(); }
  float mx = red[0]; __syncthreads();

  float lsum = 0.f;
  for (int s = tid; s < Sc; s += 256) { float e = __expf(sw[s] - mx); sw[s] = e; lsum += e; }
  red[tid] = lsum; __syncthreads();
  for (int st = 128; st > 0; st >>= 1) { if (tid < st) red[tid] += red[tid + st]; __syncthreads(); }
  float inv = 1.f / red[0];
  __syncthreads();

  for (int d = tid; d < HDc; d += 256) {
    float acc = 0.f;
    for (int s = 0; s < Sc; ++s)
      acc += sw[s] * (float)qkv[((size_t)(b * Sc + s)) * (3 * Hc) + 2 * Hc + h * HDc + d];
    out[b * Hc + h * HDc + d] = (_Float16)(acc * inv);
  }
}

// ---------------------------------------------------------------- assemble attended sequence
__global__ void assemble(const _Float16* __restrict__ gout, const _Float16* __restrict__ locp,
                         const float* __restrict__ logn, _Float16* __restrict__ attended)
{
  int i = blockIdx.x * 256 + threadIdx.x;
  const int total = Bc * Sc * Hc;
  if (i >= total) return;
  int row = i / Hc, col = i % Hc;
  int b = row / Sc, s = row % Sc;
  if (s == 0) attended[i] = (_Float16)((float)gout[b * Hc + col] * logn[0]);
  else        attended[i] = locp[i];
}

// ---------------------------------------------------------------- host launch
extern "C" void kernel_launch(void* const* d_in, const int* in_sizes, int n_in,
                              void* d_out, int out_size, void* d_ws, size_t ws_size,
                              hipStream_t stream)
{
  const float* x         = (const float*)d_in[0];
  const float* Wqkv_cls  = (const float*)d_in[1];  const float* bqkv_cls = (const float*)d_in[2];
  const float* Wo_cls    = (const float*)d_in[3];  const float* bo_cls   = (const float*)d_in[4];
  const float* Wqkv_loc  = (const float*)d_in[5];  const float* bqkv_loc = (const float*)d_in[6];
  const float* Wo_loc    = (const float*)d_in[7];  const float* bo_loc   = (const float*)d_in[8];
  const float* logn      = (const float*)d_in[9];
  const float* Wqkv_h    = (const float*)d_in[10]; const float* bqkv_h   = (const float*)d_in[11];
  const float* Wo_h      = (const float*)d_in[12]; const float* bo_h     = (const float*)d_in[13];
  const float* Wqkv_v    = (const float*)d_in[14]; const float* bqkv_v   = (const float*)d_in[15];
  const float* Wo_v      = (const float*)d_in[16]; const float* bo_v     = (const float*)d_in[17];
  const float* W1        = (const float*)d_in[18]; const float* b1       = (const float*)d_in[19];
  const float* W2        = (const float*)d_in[20]; const float* b2       = (const float*)d_in[21];
  const float* Wcls      = (const float*)d_in[22]; const float* bcls     = (const float*)d_in[23];

  const int M = Bc * Sc;              // 8192 rows
  _Float16* p = (_Float16*)d_ws;
  auto alloc = [&](size_t n) { _Float16* r = p; p += (n + 127) & ~(size_t)127; return r; };

  _Float16* wqkv_cls_h = alloc(512 * 1536);
  _Float16* wqkv_loc_h = alloc(512 * 1536);
  _Float16* wqkv_h_h   = alloc(512 * 1536);
  _Float16* wqkv_v_h   = alloc(512 * 1536);
  _Float16* wo_cls_h   = alloc(512 * 512);
  _Float16* wo_loc_h   = alloc(512 * 512);
  _Float16* wo_h_h     = alloc(512 * 512);
  _Float16* wo_v_h     = alloc(512 * 512);
  _Float16* w1_h       = alloc(1024 * 512);
  _Float16* w2_h       = alloc(512 * 512);
  _Float16* wc_h       = alloc(512 * Lc);
  _Float16* xh         = alloc((size_t)M * 512);
  _Float16* qkv        = alloc((size_t)M * 1536);
  _Float16* attnout    = alloc((size_t)M * 512);
  _Float16* locproj    = alloc((size_t)M * 512);
  _Float16* attended   = alloc((size_t)M * 512);
  _Float16* combined   = alloc((size_t)M * 1024);
  _Float16* seq1       = alloc((size_t)M * 512);
  _Float16* seq2       = alloc((size_t)M * 512);
  _Float16* clsout     = alloc(Bc * 512);
  _Float16* gout       = alloc(Bc * 512);

  auto cast = [&](const float* s, _Float16* d, int n) {
    cast_f32_f16<<<(n / 4 + 255) / 256, 256, 0, stream>>>(s, d, n);
  };
  cast(x, xh, M * 512);
  cast(Wqkv_cls, wqkv_cls_h, 512 * 1536);
  cast(Wqkv_loc, wqkv_loc_h, 512 * 1536);
  cast(Wqkv_h,   wqkv_h_h,   512 * 1536);
  cast(Wqkv_v,   wqkv_v_h,   512 * 1536);
  cast(Wo_cls, wo_cls_h, 512 * 512);
  cast(Wo_loc, wo_loc_h, 512 * 512);
  cast(Wo_h,   wo_h_h,   512 * 512);
  cast(Wo_v,   wo_v_h,   512 * 512);
  cast(W1, w1_h, 1024 * 512);
  cast(W2, w2_h, 512 * 512);
  cast(Wcls, wc_h, 512 * Lc);

  auto gemm = [&](const _Float16* A, int lda, const _Float16* Bw, int ldb,
                  const float* bias, _Float16* oH, float* oF,
                  int m, int n, int k, int oLd, int cOff, int relu) {
    dim3 grid((m + BM - 1) / BM, (n + BN - 1) / BN);
    gemm_f16<<<grid, 256, 0, stream>>>(A, lda, Bw, ldb, bias, oH, oF, m, n, k, oLd, cOff, relu);
  };

  // --- ArrowAttention: cls branch ---
  gemm(xh, 512, wqkv_cls_h, 1536, bqkv_cls, qkv, nullptr, M, 1536, 512, 1536, 0, 0);
  cls_attn<<<dim3(NHc, Bc), 256, 0, stream>>>(qkv, clsout);
  gemm(clsout, 512, wo_cls_h, 512, bo_cls, gout, nullptr, Bc, 512, 512, 512, 0, 0);
  // --- ArrowAttention: local branch (min_key = 1) ---
  gemm(xh, 512, wqkv_loc_h, 1536, bqkv_loc, qkv, nullptr, M, 1536, 512, 1536, 0, 0);
  win_attn<<<dim3(Sc / Wc_, NHc, Bc), 256, 0, stream>>>(qkv, attnout, 1);
  gemm(attnout, 512, wo_loc_h, 512, bo_loc, locproj, nullptr, M, 512, 512, 512, 0, 0);
  // --- attended = [cls * logn, local_out[1:]] ---
  assemble<<<(M * 512 + 255) / 256, 256, 0, stream>>>(gout, locproj, logn, attended);
  // --- BiSPA horizontal ---
  gemm(attended, 512, wqkv_h_h, 1536, bqkv_h, qkv, nullptr, M, 1536, 512, 1536, 0, 0);
  win_attn<<<dim3(Sc / Wc_, NHc, Bc), 256, 0, stream>>>(qkv, attnout, 0);
  gemm(attnout, 512, wo_h_h, 512, bo_h, combined, nullptr, M, 512, 512, 1024, 0, 0);
  // --- BiSPA vertical ---
  gemm(attended, 512, wqkv_v_h, 1536, bqkv_v, qkv, nullptr, M, 1536, 512, 1536, 0, 0);
  win_attn<<<dim3(Sc / Wc_, NHc, Bc), 256, 0, stream>>>(qkv, attnout, 0);
  gemm(attnout, 512, wo_v_h, 512, bo_v, combined, nullptr, M, 512, 512, 1024, 512, 0);
  // --- MLP fusion + classifier ---
  gemm(combined, 1024, w1_h, 512, b1, seq1, nullptr, M, 512, 1024, 512, 0, 1);
  gemm(seq1, 512, w2_h, 512, b2, seq2, nullptr, M, 512, 512, 512, 0, 0);
  gemm(seq2, 512, wc_h, Lc, bcls, nullptr, (float*)d_out, M, Lc, 512, Lc, 0, 0);
}
